// EMIScalePyramid_61632780698278
// MI455X (gfx1250) — matrix operand order repaired and checked by reading, not compiled, not run on hardware
//
#include <hip/hip_runtime.h>

typedef __attribute__((ext_vector_type(16))) _Float16 v16h;
typedef __attribute__((ext_vector_type(8)))  float    v8f;

// Reference constants
// ksizes = int(1.6 + 1.6*i) for i=0..3 -> {1,3,4,6}; pad = 7; BINS = 8
// 7 octaves of 1024 -> 1024,512,...,16 ; per-octave out elems = 2*4*H*W
#define PYR_TOTAL 11184128ull
#define RADII_OFF 11184128ull   // 7 * 2 * 3 = 42 floats
#define PDIST_OFF 11184170ull   // 7 * 2 * 3 = 42 floats

__global__ __launch_bounds__(32)
void emi_tile_kernel(const float* __restrict__ x, float* __restrict__ out,
                     int oct, int Ho) {
    const int lane = threadIdx.x;        // 0..31 (wave32)
    const int h    = lane >> 4;          // lane half
    const int n    = lane & 15;          // D column / A row (M)
    const int b    = blockIdx.z;
    const int i0   = blockIdx.y * 16;
    const int j0   = blockIdx.x * 16;
    const int str  = 1 << oct;

    const size_t xbase = (size_t)b * 1024u * 1024u;

    // Hint: pull the tile's first in-bounds halo row toward the caches.
    {
        int ri = i0 - 8 < 0 ? 0 : i0 - 8;
        int qi = j0 - 8 < 0 ? 0 : j0 - 8;
        __builtin_prefetch(&x[xbase + (size_t)(ri * str) * 1024u + (size_t)(qi * str)], 0, 3);
    }

    // ---- Build per-row level-set bitmasks via wave32 ballot --------------
    // Halo coords: r,q in [0,32); image pixel (i0-8+r, j0-8+q). Zero padding
    // of the reference level-sets == bit cleared for OOB lanes (all channels).
    __shared__ __align__(16) unsigned rowmask[8][32];

    for (int t = 0; t < 32; ++t) {
        int ri = i0 - 8 + t;
        int qi = j0 - 8 + lane;
        bool inb = (ri >= 0) && (ri < Ho) && (qi >= 0) && (qi < Ho);
        int idx = -1;
        if (inb) {
            float v = x[xbase + (size_t)(ri * str) * 1024u + (size_t)(qi * str)];
            idx = (int)(v * 8.0f);       // uint8 cast of x*BINS, x in [0,1)
        }
        #pragma unroll
        for (int c = 0; c < 8; ++c) {
            bool p = (c == 0) ? (idx > 0) : (idx == c);   // ch0 = ~onehot(bin0)
            unsigned m = (unsigned)__ballot(p ? 1 : 0);   // bit l == halo col l
            if (lane == c) rowmask[c][t] = m;
        }
    }
    __syncthreads();

    // ---- Constant per-lane data ------------------------------------------
    const int kk[4] = {1, 3, 4, 6};

    // Vertical window mask as WMMA A-matrix (16x32 f16).
    // A layout: lane half h, M = lane%16; element e -> K = 16h+e.
    // Window over halo rows r: m+9-k <= r <= m+8+k  (rows i-k+1..i+k, halo shift +8).
    v16h amat[4];
    #pragma unroll
    for (int ki = 0; ki < 4; ++ki) {
        const int k = kk[ki];
        v16h a;
        #pragma unroll
        for (int e = 0; e < 16; ++e) {
            int r = 16 * h + e;
            bool inw = (r >= n + 9 - k) && (r <= n + 8 + k);
            a[e] = inw ? (_Float16)1.0f : (_Float16)0.0f;
        }
        amat[ki] = a;
    }

    // Horizontal window bitmask over halo cols q: n+9-k .. n+8+k (2k bits).
    unsigned wm[4];
    #pragma unroll
    for (int ki = 0; ki < 4; ++ki) {
        const int k = kk[ki];
        wm[ki] = ((1u << (2 * k)) - 1u) << (n + 9 - k);
    }

    // ---- Box counts via WMMA, entropy accumulation ------------------------
    v8f rc0[4];          // 1 / max(count_ch0, 1) per scale
    v8f mi[4] = {};      // sum an*log(an)

    #pragma unroll
    for (int c = 0; c < 8; ++c) {
        // 16 row masks for this lane's K rows: rowmask[c][16h .. 16h+15]
        const uint4* mp = (const uint4*)(&rowmask[c][h * 16]);
        uint4 qa = mp[0], qb = mp[1], qc = mp[2], qd = mp[3];
        unsigned mr[16] = {qa.x, qa.y, qa.z, qa.w,  qb.x, qb.y, qb.z, qb.w,
                           qc.x, qc.y, qc.z, qc.w,  qd.x, qd.y, qd.z, qd.w};

        #pragma unroll
        for (int ki = 0; ki < 4; ++ki) {
            // B-matrix (32x16 f16): element e -> K = 16h+e, N = lane%16;
            // value = horizontal window count at halo row 16h+e, col n (<= 12, exact f16)
            v16h bmat;
            #pragma unroll
            for (int e = 0; e < 16; ++e)
                bmat[e] = (_Float16)__popc(mr[e] & wm[ki]);

            v8f cz = {};
            // counts(16x16) = Vmask(16x32) x Hsums(32x16), f32 accumulate (exact ints)
            v8f S = __builtin_amdgcn_wmma_f32_16x16x32_f16(
                false, amat[ki], false, bmat, (short)0, cz, false, false);

            if (c == 0) {
                #pragma unroll
                for (int g = 0; g < 8; ++g)
                    rc0[ki][g] = 1.0f / fmaxf(S[g], 1.0f);
            } else {
                #pragma unroll
                for (int g = 0; g < 8; ++g) {
                    float a  = fmaxf(S[g], 1.0f);
                    float an = a * rc0[ki][g];
                    mi[ki][g] += an * __logf(an);
                }
            }
        }
    }

    // ---- Store: out[b, 0, ki, i, j] = -mi ; D layout: M=g+8h, N=lane%16 ----
    #pragma unroll
    for (int ki = 0; ki < 4; ++ki) {
        #pragma unroll
        for (int g = 0; g < 8; ++g) {
            int i = i0 + g + 8 * h;
            int j = j0 + n;
            out[((size_t)(b * 4 + ki) * Ho + i) * (size_t)Ho + j] = -mi[ki][g];
        }
    }
}

__global__ void emi_tail_kernel(float* __restrict__ out) {
    int t = threadIdx.x;
    if (t < 42) {
        int j = t % 3;                                   // 2 * radii[:-1]
        out[RADII_OFF + t] = (float)(2.0 * (1.6 + 1.6 * (double)j));
    } else if (t < 84) {
        int u = t - 42;
        int o = u / 6;                                   // 2^octave
        out[PDIST_OFF + u] = (float)(1 << o);
    }
}

extern "C" void kernel_launch(void* const* d_in, const int* in_sizes, int n_in,
                              void* d_out, int out_size, void* d_ws, size_t ws_size,
                              hipStream_t stream) {
    const float* x = (const float*)d_in[0];
    float* out = (float*)d_out;

    size_t base = 0;
    for (int o = 0; o < 7; ++o) {
        int Ho = 1024 >> o;
        dim3 grid(Ho / 16, Ho / 16, 2);
        emi_tile_kernel<<<grid, dim3(32, 1, 1), 0, stream>>>(x, out + base, o, Ho);
        base += (size_t)8 * Ho * Ho;   // 2 batches * 4 scales * Ho * Ho
    }
    emi_tail_kernel<<<1, 96, 0, stream>>>(out);
}
